// TextGuidedSampler_80298708566539
// MI455X (gfx1250) — matrix operand order repaired and checked by reading, not compiled, not run on hardware
//
#include <hip/hip_runtime.h>
#include <math.h>

typedef __attribute__((ext_vector_type(16))) _Float16 v16h;
typedef __attribute__((ext_vector_type(8)))  float    v8f;

#define N_IMG 4608
#define N_TXT 128
#define DIM   4096
#define TOPP  0.9f

#define CVT4(dst, i, v)                         \
  {                                             \
    dst[(i) + 0] = (_Float16)(v).x;             \
    dst[(i) + 1] = (_Float16)(v).y;             \
    dst[(i) + 2] = (_Float16)(v).z;             \
    dst[(i) + 3] = (_Float16)(v).w;             \
  }

#define SSQ4(acc, v) \
  { acc += (v).x * (v).x + (v).y * (v).y + (v).z * (v).z + (v).w * (v).w; }

// ---------------------------------------------------------------------------
// K1: per-text-token scale s_t = mask[t] ? 1/max(||text_t||, eps) : 0
// ---------------------------------------------------------------------------
__global__ __launch_bounds__(256) void k1_text_scale(
    const float* __restrict__ T, const int* __restrict__ mask,
    float* __restrict__ sscale) {
  int t = blockIdx.x;                     // 0..127
  const float* row = T + (size_t)t * DIM;
  float s = 0.f;
  for (int d = threadIdx.x; d < DIM; d += 256) {
    float v = row[d];
    s = fmaf(v, v, s);
  }
  __shared__ float red[256];
  red[threadIdx.x] = s;
  __syncthreads();
  for (int o = 128; o > 0; o >>= 1) {
    if (threadIdx.x < o) red[threadIdx.x] += red[threadIdx.x + o];
    __syncthreads();
  }
  if (threadIdx.x == 0) {
    float nrm = fmaxf(sqrtf(red[0]), 1e-8f);
    sscale[t] = (mask[t] != 0) ? (1.0f / nrm) : 0.0f;
  }
}

// ---------------------------------------------------------------------------
// K2: WMMA f16 GEMM (16-row tile x all 128 text cols, K=4096), fold in the
// per-column scale, reduce over columns, divide by image-row norm.
// 8 waves/block; wave w owns N-tile w. K unrolled x2 -> 2 wmma per trip,
// running pointers so all 16 b128 loads fold into immediate offsets.
// ---------------------------------------------------------------------------
__global__ __launch_bounds__(256) void k2_sim_wmma(
    const float* __restrict__ A,       // local_f   [4608, 4096]
    const float* __restrict__ T,       // text_emb  [128, 4096]
    const float* __restrict__ sscale,  // [128]
    float* __restrict__ sim_sum) {     // [4608]
  const int wave = threadIdx.x >> 5;
  const int lane = threadIdx.x & 31;
  const int m    = lane & 15;          // row within 16-row tile / N within tile
  const int h    = lane >> 4;          // lane-half selector
  const int grow = blockIdx.x * 16 + m;        // global A row
  const int n    = wave * 16 + m;              // text row (= N index)

  // A lane covers K = {8h..8h+7, 16+8h..23+8h} per 32-chunk; fold 8h into base.
  const float* ap = A + (size_t)grow * DIM + h * 8;
  // B lane covers K = {8h..8h+7, 16+8h..23+8h} per 32-chunk; fold 8h into base.
  const float* tp = T + (size_t)n * DIM + h * 8;

  v8f acc = {};
  float ssq = 0.f;

  for (int k0 = 0; k0 < DIM; k0 += 64) {
    float4 a0 = *(const float4*)(ap + 0);
    float4 a1 = *(const float4*)(ap + 4);
    float4 a2 = *(const float4*)(ap + 16);
    float4 a3 = *(const float4*)(ap + 20);
    float4 a4 = *(const float4*)(ap + 32);
    float4 a5 = *(const float4*)(ap + 36);
    float4 a6 = *(const float4*)(ap + 48);
    float4 a7 = *(const float4*)(ap + 52);

    float4 b0 = *(const float4*)(tp + 0);
    float4 b1 = *(const float4*)(tp + 4);
    float4 b2 = *(const float4*)(tp + 16);
    float4 b3 = *(const float4*)(tp + 20);
    float4 b4 = *(const float4*)(tp + 32);
    float4 b5 = *(const float4*)(tp + 36);
    float4 b6 = *(const float4*)(tp + 48);
    float4 b7 = *(const float4*)(tp + 52);

    v16h av0, bv0, av1, bv1;
    CVT4(av0, 0, a0); CVT4(av0, 4, a1); CVT4(av0, 8, a2); CVT4(av0, 12, a3);
    CVT4(bv0, 0, b0); CVT4(bv0, 4, b1); CVT4(bv0, 8, b2); CVT4(bv0, 12, b3);
    CVT4(av1, 0, a4); CVT4(av1, 4, a5); CVT4(av1, 8, a6); CVT4(av1, 12, a7);
    CVT4(bv1, 0, b4); CVT4(bv1, 4, b5); CVT4(bv1, 8, b6); CVT4(bv1, 12, b7);

    if (wave == 0) {  // wave-uniform; lanes m and m+16 together cover all K
      SSQ4(ssq, a0); SSQ4(ssq, a1); SSQ4(ssq, a2); SSQ4(ssq, a3);
      SSQ4(ssq, a4); SSQ4(ssq, a5); SSQ4(ssq, a6); SSQ4(ssq, a7);
    }

    acc = __builtin_amdgcn_wmma_f32_16x16x32_f16(
        false, av0, false, bv0, (short)0, acc, false, false);
    acc = __builtin_amdgcn_wmma_f32_16x16x32_f16(
        false, av1, false, bv1, (short)0, acc, false, false);

    ap += 64;
    tp += 64;
  }

  __shared__ float ldsC[16 * 132];
  __shared__ float ldsS[32];

  float sc = sscale[n];   // lane's N column scale (mask / txt_norm)
#pragma unroll
  for (int r = 0; r < 8; ++r) {
    int M = r + 8 * h;    // C layout: VGPR r -> M=r (lanes 0-15), M=r+8 (16-31)
    ldsC[M * 132 + wave * 16 + m] = acc[r] * sc;
  }
  if (wave == 0) ldsS[lane] = ssq;
  __syncthreads();

  if (threadIdx.x < 16) {
    int M = threadIdx.x;
    float s = 0.f;
    for (int nn = 0; nn < N_TXT; ++nn) s += ldsC[M * 132 + nn];
    float nrm = sqrtf(ldsS[M] + ldsS[M + 16]);
    sim_sum[blockIdx.x * 16 + M] = s / fmaxf(nrm, 1e-8f);
  }
}

// ---------------------------------------------------------------------------
// K3: softmax(sim*2) -> +gumbel -> softmax   (single block, LDS reductions)
// ---------------------------------------------------------------------------
__global__ __launch_bounds__(256) void k3_softmax_chain(
    const float* __restrict__ sim, const float* __restrict__ g,
    float* __restrict__ p1tmp, float* __restrict__ probs) {
  __shared__ float red[256];
  const int tid = threadIdx.x;

  float m = -INFINITY;
  for (int i = tid; i < N_IMG; i += 256) m = fmaxf(m, sim[i] * 2.0f);
  red[tid] = m; __syncthreads();
  for (int o = 128; o > 0; o >>= 1) {
    if (tid < o) red[tid] = fmaxf(red[tid], red[tid + o]);
    __syncthreads();
  }
  float m1 = red[0]; __syncthreads();

  float z = 0.f;
  for (int i = tid; i < N_IMG; i += 256) z += expf(sim[i] * 2.0f - m1);
  red[tid] = z; __syncthreads();
  for (int o = 128; o > 0; o >>= 1) {
    if (tid < o) red[tid] += red[tid + o];
    __syncthreads();
  }
  float Z1 = red[0]; __syncthreads();

  float m2 = -INFINITY;
  for (int i = tid; i < N_IMG; i += 256) {
    float p = expf(sim[i] * 2.0f - m1) / Z1 + g[i];
    p1tmp[i] = p;
    m2 = fmaxf(m2, p);
  }
  red[tid] = m2; __syncthreads();
  for (int o = 128; o > 0; o >>= 1) {
    if (tid < o) red[tid] = fmaxf(red[tid], red[tid + o]);
    __syncthreads();
  }
  float M2 = red[0]; __syncthreads();

  float z2 = 0.f;
  for (int i = tid; i < N_IMG; i += 256) z2 += expf(p1tmp[i] - M2);
  red[tid] = z2; __syncthreads();
  for (int o = 128; o > 0; o >>= 1) {
    if (tid < o) red[tid] += red[tid + o];
    __syncthreads();
  }
  float Z2 = red[0]; __syncthreads();

  for (int i = tid; i < N_IMG; i += 256) probs[i] = expf(p1tmp[i] - M2) / Z2;
}

// ---------------------------------------------------------------------------
// K4: stable descending rank of each prob (== argsort(-p) position) + scatter
// the value to its sorted slot. O(N^2) compares, all L2-resident.
// ---------------------------------------------------------------------------
__global__ __launch_bounds__(256) void k4_rank(
    const float* __restrict__ probs, int* __restrict__ ranks,
    float* __restrict__ sorted) {
  const int i = blockIdx.x;
  const float pi = probs[i];
  const int tid = threadIdx.x;
  int c = 0;
  for (int j = tid; j < N_IMG; j += 256) {
    float pj = probs[j];
    c += (pj > pi) || (pj == pi && j < i);
  }
  __shared__ int red[256];
  red[tid] = c; __syncthreads();
  for (int o = 128; o > 0; o >>= 1) {
    if (tid < o) red[tid] += red[tid + o];
    __syncthreads();
  }
  if (tid == 0) {
    ranks[i] = red[0];
    sorted[red[0]] = pi;
  }
}

// ---------------------------------------------------------------------------
// K5: single-block inclusive scan over sorted probs -> k = min(#(cum<=p)+1, N)
// ---------------------------------------------------------------------------
__global__ __launch_bounds__(256) void k5_scan(
    const float* __restrict__ sorted, int* __restrict__ kout) {
  __shared__ float buf[256];
  __shared__ float carry_s;
  __shared__ int cnt_s;
  const int tid = threadIdx.x;
  if (tid == 0) { carry_s = 0.f; cnt_s = 0; }
  __syncthreads();
  for (int c = 0; c < N_IMG / 256; ++c) {
    buf[tid] = sorted[c * 256 + tid];
    __syncthreads();
    for (int o = 1; o < 256; o <<= 1) {
      float add = (tid >= o) ? buf[tid - o] : 0.f;
      __syncthreads();
      buf[tid] += add;
      __syncthreads();
    }
    float cum = buf[tid] + carry_s;
    if (cum <= TOPP) atomicAdd(&cnt_s, 1);
    __syncthreads();
    if (tid == 255) carry_s += buf[255];
    __syncthreads();
  }
  if (tid == 0) {
    int k = cnt_s + 1;
    if (k > N_IMG) k = N_IMG;
    *kout = k;
  }
}

// ---------------------------------------------------------------------------
// K6: out = local_f * (rank < k), float4-vectorized streaming pass
// ---------------------------------------------------------------------------
__global__ __launch_bounds__(256) void k6_output(
    const float4* __restrict__ A, const int* __restrict__ ranks,
    const int* __restrict__ kptr, float4* __restrict__ out) {
  size_t idx = (size_t)blockIdx.x * 256 + threadIdx.x;  // float4 index
  int row = (int)(idx >> 10);                           // DIM/4 = 1024 f4/row
  float keep = (ranks[row] < *kptr) ? 1.0f : 0.0f;
  float4 v = A[idx];
  v.x *= keep; v.y *= keep; v.z *= keep; v.w *= keep;
  out[idx] = v;
}

// ---------------------------------------------------------------------------
extern "C" void kernel_launch(void* const* d_in, const int* in_sizes, int n_in,
                              void* d_out, int out_size, void* d_ws, size_t ws_size,
                              hipStream_t stream) {
  const float* local_f = (const float*)d_in[0];   // [4608,4096] f32
  const float* text    = (const float*)d_in[1];   // [128,4096]  f32
  const int*   mask    = (const int*)  d_in[2];   // [128] bool->int
  const float* gumbel  = (const float*)d_in[3];   // [4608] f32
  float* out = (float*)d_out;

  // scratch layout (~93 KB, lives in L2)
  float* wsf     = (float*)d_ws;
  float* sscale  = wsf;                          // 128
  float* sim_sum = wsf + 128;                    // 4608
  float* p1tmp   = sim_sum + N_IMG;              // 4608
  float* probs   = p1tmp + N_IMG;                // 4608
  float* sorted  = probs + N_IMG;                // 4608
  int*   ranks   = (int*)(sorted + N_IMG);       // 4608
  int*   kout    = ranks + N_IMG;                // 1

  k1_text_scale<<<N_TXT, 256, 0, stream>>>(text, mask, sscale);
  k2_sim_wmma<<<N_IMG / 16, 256, 0, stream>>>(local_f, text, sscale, sim_sum);
  k3_softmax_chain<<<1, 256, 0, stream>>>(sim_sum, gumbel, p1tmp, probs);
  k4_rank<<<N_IMG, 256, 0, stream>>>(probs, ranks, sorted);
  k5_scan<<<1, 256, 0, stream>>>(sorted, kout);
  k6_output<<<(N_IMG * (DIM / 4)) / 256, 256, 0, stream>>>(
      (const float4*)local_f, ranks, kout, (float4*)out);
}
